// WindowAttentionAGA_74423193305339
// MI455X (gfx1250) — matrix operand order repaired and checked by reading, not compiled, not run on hardware
//
#include <hip/hip_runtime.h>

typedef __attribute__((ext_vector_type(16))) _Float16    v16h;
typedef __attribute__((ext_vector_type(8)))  float       v8f;
typedef __attribute__((ext_vector_type(4)))  unsigned int u32x4;

#define DIMC 180
#define NHEADS 6
#define HD 30
#define HDP 32
#define NTOK 64
#define KDIM 16
#define QSCALE 0.18257418583505536f  /* 30^-0.5 */

constexpr int QKV_NT = 34;    // ceil(540/16) -> padded 544
constexpr int QKV_KT = 6;     // 192/32
constexpr int PROJ_NT = 12;   // 192/16
constexpr int PROJ_KT = 6;    // 192/32
constexpr int QKVP_N  = QKV_KT * QKV_NT * 32 * 16;   // 104448 halves
constexpr int PROJP_N = PROJ_KT * PROJ_NT * 32 * 16; //  36864 halves

// ---- LDS layout (bytes) ----
constexpr int XH_OFF  = 0;       // [64][192] f16 : x (f16)  -- later aliased as attn-out
constexpr int QH_OFF  = 24576;   // [6][64][32] f16 : Q (pre-scaled)
constexpr int KH_OFF  = 49152;   // [6][64][32] f16 : K
constexpr int VT_OFF  = 73728;   // [6][32][64] f16 : V transposed (hd-major)
constexpr int DB_OFF  = 98304;   // [64][64] f32 : -||bpe_i - bpe_j||
constexpr int SB_OFF  = 114688;  // [64][64] f32 : stage A: bpe staging; stage B: epilogue LUT; stage C: scores
constexpr int PH_OFF  = 131072;  // [64][64] f16 : exp(P) unnormalized
constexpr int RED_OFF = 139264;  // redmax[256] f32, redsum[256] f32, rinv[64] f32
constexpr int LDS_BYTES = RED_OFF + 256*4 + 256*4 + 64*4; // 141568 (< 160 KB)

__device__ __forceinline__ v16h load_split(const _Float16* p0, const _Float16* p1) {
  union { u32x4 q[2]; v16h h; } u;
  u.q[0] = *(const u32x4*)p0;   // halves j=0..7
  u.q[1] = *(const u32x4*)p1;   // halves j=8..15
  return u.h;
}
__device__ __forceinline__ v16h load_contig(const _Float16* p) {
  return load_split(p, p + 8);
}

// ---------------------------------------------------------------------------
// Pre-pass: convert + pad + swizzle weights into wave32 WMMA B-operand layout.
// B fragment in this code's contiguous k order (same as A): lane L holds output column n = L%16;
// half j holds K = j + 16*(L/16). Packed so each lane loads 32 contiguous B.
// ---------------------------------------------------------------------------
__global__ void pack_weights_kernel(const float* __restrict__ qkv_w,
                                    const float* __restrict__ proj_w,
                                    _Float16* __restrict__ qkvP,
                                    _Float16* __restrict__ projP) {
  int idx = blockIdx.x * 256 + threadIdx.x;
  if (idx < QKVP_N) {
    int j = idx & 15, L = (idx >> 4) & 31, tile = idx >> 9;
    int nt = tile % QKV_NT, kt = tile / QKV_NT;   // packed index = (kt*QKV_NT+nt)
    int n = nt * 16 + (L & 15);
    int k = kt * 32 + j + ((L >> 4) << 4);
    float v = (n < 3 * DIMC && k < DIMC) ? qkv_w[n * DIMC + k] : 0.f;
    qkvP[idx] = (_Float16)v;
  } else {
    int i2 = idx - QKVP_N;
    if (i2 < PROJP_N) {
      int j = i2 & 15, L = (i2 >> 4) & 31, tile = i2 >> 9;
      int nt = tile % PROJ_NT, kt = tile / PROJ_NT;
      int n = nt * 16 + (L & 15);
      int k = kt * 32 + j + ((L >> 4) << 4);
      float v = (n < DIMC && k < DIMC) ? proj_w[n * DIMC + k] : 0.f;
      projP[i2] = (_Float16)v;
    }
  }
}

// ---------------------------------------------------------------------------
// Main kernel: one 8-wave workgroup per window.
// ---------------------------------------------------------------------------
__global__ __launch_bounds__(256)
void winattn_kernel(const float* __restrict__ x, const float* __restrict__ bpe,
                    const _Float16* __restrict__ qkvP, const float* __restrict__ qkv_b,
                    const _Float16* __restrict__ projP, const float* __restrict__ proj_b,
                    const float* __restrict__ tau, float* __restrict__ out) {
  __shared__ __align__(16) char smem[LDS_BYTES];
  _Float16* xh = (_Float16*)(smem + XH_OFF);
  _Float16* qh = (_Float16*)(smem + QH_OFF);
  _Float16* kh = (_Float16*)(smem + KH_OFF);
  _Float16* vT = (_Float16*)(smem + VT_OFF);
  float*    dB = (float*)(smem + DB_OFF);
  float*    sb = (float*)(smem + SB_OFF);
  float*    bpeL = (float*)(smem + SB_OFF);      // alias (stage A only)
  int4*     lut  = (int4*)(smem + SB_OFF);       // alias (stage B only)
  _Float16* ph = (_Float16*)(smem + PH_OFF);
  float*    redmax = (float*)(smem + RED_OFF);
  float*    redsum = redmax + 256;
  float*    rinv   = redsum + 256;

  const int tid  = threadIdx.x;
  const int lane = tid & 31;
  const int l16  = lane & 15;
  const int hi   = (lane >> 4) & 1;
  const int wid  = tid >> 5;
  const int b    = blockIdx.x;

  // ---- Stage A: zero padded buffers, stage bpe, convert x, distances ----
  {
    const u32x4 zv = {};
    u32x4* z = (u32x4*)(smem + QH_OFF);          // zero Q/K/V (covers hd 30..31 padding)
    for (int i = tid; i < (24576 * 3) / 16; i += 256) z[i] = zv;
    u32x4* zx = (u32x4*)(smem + XH_OFF);         // zero x (covers cols 180..191 padding)
    for (int i = tid; i < 24576 / 16; i += 256) zx[i] = zv;
  }
  const float* bpeG = bpe + (size_t)b * (NTOK * KDIM);
  for (int i = tid; i < (NTOK * KDIM) / 4; i += 256)
    ((float4*)bpeL)[i] = ((const float4*)bpeG)[i];
  __syncthreads();

  // x -> f16 LDS (vectorized: 180 % 4 == 0, each float4 stays within a row)
  const float* xb = x + (size_t)b * (NTOK * DIMC);
  for (int i = tid; i < (NTOK * DIMC) / 4; i += 256) {
    const int m = i / 45, c4 = (i - m * 45) * 4;
    float4 xv = *(const float4*)(xb + i * 4);
    union { _Float16 h[4]; uint2 u; } cv;
    cv.h[0] = (_Float16)xv.x; cv.h[1] = (_Float16)xv.y;
    cv.h[2] = (_Float16)xv.z; cv.h[3] = (_Float16)xv.w;
    *(uint2*)(xh + m * 192 + c4) = cv.u;
  }
  // pairwise -||bpe_i - bpe_j||: thread owns row i = tid/4, 16 cols
  {
    const int i = tid >> 2;
    const int j0 = (tid & 3) * 16;
    const float4* qi = (const float4*)(bpeL + i * KDIM);
    const float4 q0 = qi[0], q1 = qi[1], q2 = qi[2], q3 = qi[3];
    for (int jj = 0; jj < 16; ++jj) {
      const int j = j0 + jj;
      const float4* bj = (const float4*)(bpeL + j * KDIM);
      const float4 b0 = bj[0], b1 = bj[1], b2 = bj[2], b3 = bj[3];
      float s = 0.f, d;
      d = q0.x - b0.x; s += d * d;  d = q0.y - b0.y; s += d * d;
      d = q0.z - b0.z; s += d * d;  d = q0.w - b0.w; s += d * d;
      d = q1.x - b1.x; s += d * d;  d = q1.y - b1.y; s += d * d;
      d = q1.z - b1.z; s += d * d;  d = q1.w - b1.w; s += d * d;
      d = q2.x - b2.x; s += d * d;  d = q2.y - b2.y; s += d * d;
      d = q2.z - b2.z; s += d * d;  d = q2.w - b2.w; s += d * d;
      d = q3.x - b3.x; s += d * d;  d = q3.y - b3.y; s += d * d;
      d = q3.z - b3.z; s += d * d;  d = q3.w - b3.w; s += d * d;
      dB[i * 64 + j] = -sqrtf(s);
    }
  }
  __syncthreads();

  // Build stage-B epilogue LUT (one int4 per output feature; kills per-tile div/mod)
  for (int f = tid; f < QKV_NT * 16; f += 256) {
    int4 d;
    if (f < 3 * DIMC) {
      const int which = f / DIMC;
      const int fr = f - which * DIMC;
      const int head = fr / HD;
      const int hd = fr - head * HD;
      if (which == 0)      { d.x = QH_OFF + head * 4096 + hd * 2;   d.y = 64; d.z = __float_as_int(QSCALE); }
      else if (which == 1) { d.x = KH_OFF + head * 4096 + hd * 2;   d.y = 64; d.z = __float_as_int(1.0f); }
      else                 { d.x = VT_OFF + head * 4096 + hd * 128; d.y = 2;  d.z = __float_as_int(1.0f); }
      d.w = __float_as_int(qkv_b[f]);
    } else {
      d.x = -1; d.y = 0; d.z = 0; d.w = 0;
    }
    lut[f] = d;
  }
  __syncthreads();

  // ---- Stage B: QKV projection (M=64, K=192, N=544 padded) ----
  // wave -> fixed mt = wid/2; nt-range = (wid&1)*17 .. +16 ; A frags cached in regs
  {
    const int mtB = wid >> 1;
    const int ntBase = (wid & 1) * 17;
    const int mRow = mtB * 16 + l16;
    v16h af[QKV_KT];
#pragma unroll
    for (int kt = 0; kt < QKV_KT; ++kt) {
      const _Float16* a0 = xh + mRow * 192 + kt * 32 + 16 * hi;
      af[kt] = load_contig(a0);
    }
    for (int nn = 0; nn < 17; ++nn) {
      const int nt = ntBase + nn;
      v8f acc = {};
#pragma unroll
      for (int kt = 0; kt < QKV_KT; ++kt) {
        v16h bf = load_contig(qkvP + ((size_t)(kt * QKV_NT + nt) * 32 + lane) * 16);
        acc = __builtin_amdgcn_wmma_f32_16x16x32_f16(false, af[kt], false, bf, (short)0, acc, false, false);
      }
      const int4 d = lut[nt * 16 + l16];
      if (d.x >= 0) {
        const float mul  = __int_as_float(d.z);
        const float bias = __int_as_float(d.w);
        char* p = smem + d.x + (mtB * 16 + 8 * hi) * d.y;
#pragma unroll
        for (int r = 0; r < 8; ++r)
          *(_Float16*)(p + r * d.y) = (_Float16)((acc[r] + bias) * mul);
      }
    }
  }
  __syncthreads();

  // zero attn-out buffer (aliases xh; covers cols 180..191 padding for proj GEMM)
  {
    const u32x4 zv = {};
    u32x4* z = (u32x4*)(smem + XH_OFF);
    for (int i = tid; i < 24576 / 16; i += 256) z[i] = zv;
  }
  __syncthreads();
  _Float16* oh = xh;

  // ---- Stage C: per-head attention ----
  for (int h = 0; h < NHEADS; ++h) {
    const float itau = 1.f / fmaxf(tau[h], 1e-6f);
    const _Float16* qhh = qh + h * NTOK * HDP;
    const _Float16* khh = kh + h * NTOK * HDP;
    const _Float16* vhh = vT + h * HDP * NTOK;

    // S = Q@K^T + bias   (2 tiles per wave, K=32 in a single WMMA)
    for (int tt = wid; tt < 16; tt += 8) {
      const int mt = tt >> 2, nt = tt & 3;
      const int m = mt * 16 + l16;
      const int n = nt * 16 + l16;
      const _Float16* a0 = qhh + m * HDP + 16 * hi;
      v16h a = load_contig(a0);
      v16h bf = load_contig(khh + n * HDP + 16 * hi);
      v8f acc = {};
      acc = __builtin_amdgcn_wmma_f32_16x16x32_f16(false, a, false, bf, (short)0, acc, false, false);
#pragma unroll
      for (int r = 0; r < 8; ++r) {
        const int mm = mt * 16 + r + 8 * hi;
        sb[mm * 64 + n] = acc[r] + dB[mm * 64 + n] * itau;
      }
    }
    __syncthreads();

    // softmax: 4 threads per row; fold 1/rowsum into P@V epilogue
    {
      const int row = tid >> 2, ch = tid & 3;
      const float* srow = sb + row * 64 + ch * 16;
      float v[16];
      float mx = -3.4e38f;
#pragma unroll
      for (int q = 0; q < 16; ++q) { v[q] = srow[q]; mx = fmaxf(mx, v[q]); }
      redmax[tid] = mx;
      __syncthreads();
      const float* rm4 = redmax + row * 4;
      const float rm = fmaxf(fmaxf(rm4[0], rm4[1]), fmaxf(rm4[2], rm4[3]));
      float ps = 0.f;
      _Float16* prow = ph + row * 64 + ch * 16;
#pragma unroll
      for (int q = 0; q < 16; ++q) {
        float e = __expf(v[q] - rm);
        ps += e;
        prow[q] = (_Float16)e;
      }
      redsum[tid] = ps;
      __syncthreads();
      if (tid < 64) {
        const float* rs4 = redsum + tid * 4;
        rinv[tid] = 1.f / (rs4[0] + rs4[1] + rs4[2] + rs4[3]);
      }
      __syncthreads();
    }

    // O_h = softmax(P)@V   (1 tile per wave: mt=wid/2, nt=wid%2; K=64 -> 2 WMMAs)
    {
      const int mt = wid >> 1, nt = wid & 1;
      const int m = mt * 16 + l16;
      const int n = nt * 16 + l16;  // hd column
      v8f acc = {};
#pragma unroll
      for (int kt = 0; kt < 2; ++kt) {
        const _Float16* a0 = ph + m * 64 + kt * 32 + 16 * hi;
        v16h a = load_contig(a0);
        v16h bf = load_contig(vhh + n * NTOK + kt * 32 + 16 * hi);
        acc = __builtin_amdgcn_wmma_f32_16x16x32_f16(false, a, false, bf, (short)0, acc, false, false);
      }
      if (n < HD) {
#pragma unroll
        for (int r = 0; r < 8; ++r) {
          const int mm = mt * 16 + r + 8 * hi;
          oh[mm * 192 + h * HD + n] = (_Float16)(acc[r] * rinv[mm]);
        }
      }
    }
    __syncthreads();
  }

  // ---- Stage D: output projection (M=64, K=192, N=192 padded) ----
  // wave -> fixed mt = wid/2; nt-range = (wid&1)*6 .. +5 ; A frags cached in regs
  {
    float* outb = out + (size_t)b * (NTOK * DIMC);
    const int mtD = wid >> 1;
    const int ntBase = (wid & 1) * 6;
    const int mRow = mtD * 16 + l16;
    v16h ad[PROJ_KT];
#pragma unroll
    for (int kt = 0; kt < PROJ_KT; ++kt) {
      const _Float16* a0 = oh + mRow * 192 + kt * 32 + 16 * hi;
      ad[kt] = load_contig(a0);
    }
    for (int nn = 0; nn < 6; ++nn) {
      const int nt = ntBase + nn;
      v8f acc = {};
#pragma unroll
      for (int kt = 0; kt < PROJ_KT; ++kt) {
        v16h bf = load_contig(projP + ((size_t)(kt * PROJ_NT + nt) * 32 + lane) * 16);
        acc = __builtin_amdgcn_wmma_f32_16x16x32_f16(false, ad[kt], false, bf, (short)0, acc, false, false);
      }
      const int f = nt * 16 + l16;
      if (f < DIMC) {
        const float bias = proj_b[f];
        float* po = outb + (mtD * 16 + 8 * hi) * DIMC + f;
#pragma unroll
        for (int r = 0; r < 8; ++r) po[r * DIMC] = acc[r] + bias;
      }
    }
  }
}

extern "C" void kernel_launch(void* const* d_in, const int* in_sizes, int n_in,
                              void* d_out, int out_size, void* d_ws, size_t ws_size,
                              hipStream_t stream) {
  (void)n_in; (void)out_size; (void)ws_size;
  const float* x      = (const float*)d_in[0];
  const float* bpe    = (const float*)d_in[1];
  const float* qkv_w  = (const float*)d_in[2];
  const float* qkv_b  = (const float*)d_in[3];
  const float* proj_w = (const float*)d_in[4];
  const float* proj_b = (const float*)d_in[5];
  const float* tau    = (const float*)d_in[6];
  float* out = (float*)d_out;

  _Float16* qkvP  = (_Float16*)d_ws;
  _Float16* projP = qkvP + QKVP_N;

  const int packN = QKVP_N + PROJP_N;
  pack_weights_kernel<<<(packN + 255) / 256, 256, 0, stream>>>(qkv_w, proj_w, qkvP, projP);

  const int nWin = in_sizes[0] / (NTOK * DIMC);  // 8192
  winattn_kernel<<<nWin, 256, 0, stream>>>(x, bpe, qkvP, qkv_b, projP, proj_b, tau, out);
}
